// Model_37520834298266
// MI455X (gfx1250) — matrix operand order repaired and checked by reading, not compiled, not run on hardware
//
#include <hip/hip_runtime.h>
#include <hip/hip_bf16.h>
#include <math.h>

// ---------------------------------------------------------------------------
// CDNA5 (gfx1250) capsule-EM pipeline.
//  conv(+BN,+sigmoid)  : v_wmma_f32_16x16x32_bf16 implicit GEMM
//  votes (4x4 matmuls) : v_wmma_f32_16x16x4_f32 (K=4 exact match)
//  EM routing          : per-site workgroup, votes cached in LDS (bf16),
//                        staged via global_load_async_to_lds_b128 (ASYNCcnt)
// ---------------------------------------------------------------------------

typedef __attribute__((ext_vector_type(16))) __bf16          v16bf;
typedef __attribute__((ext_vector_type(16))) unsigned short  v16us;
typedef __attribute__((ext_vector_type(8)))  float           v8f;
typedef __attribute__((ext_vector_type(2)))  float           v2f;

#if defined(__HIP_DEVICE_COMPILE__) && \
    __has_builtin(__builtin_amdgcn_global_load_async_to_lds_b128)
#define HAVE_ASYNC_LDS 1
#else
#define HAVE_ASYNC_LDS 0
#endif

#if HAVE_ASYNC_LDS
// builtin signature (from compiler diagnostic): arg0 = int4* in global AS,
// arg1 = LDS side, arg2/arg3 = immediate offset / cpol
typedef int async_v4i __attribute__((vector_size(16)));
typedef __attribute__((address_space(1))) async_v4i* glb_v4i_ptr;
typedef __attribute__((address_space(3))) async_v4i* lds_v4i_ptr;
#endif

__device__ __forceinline__ unsigned short f32_to_bf16_bits(float f) {
  union { float f; unsigned u; } cv; cv.f = f;
  unsigned u = cv.u;
  unsigned r = u + 0x7FFFu + ((u >> 16) & 1u);   // round-to-nearest-even
  return (unsigned short)(r >> 16);
}
__device__ __forceinline__ float bf16_bits_to_f32(unsigned short h) {
  union { unsigned u; float f; } cv; cv.u = ((unsigned)h) << 16;
  return cv.f;
}

__device__ __forceinline__ void wait_asynccnt_zero() {
#if defined(__HIP_DEVICE_COMPILE__)
#if __has_builtin(__builtin_amdgcn_s_wait_asynccnt)
  __builtin_amdgcn_s_wait_asynccnt(0);
#else
  asm volatile("s_wait_asynccnt 0x0" ::: "memory");
#endif
#endif
}

// ---------------------------------------------------------------------------
// Kernel 1: 3x3 conv (pad=1) + BN (+sigmoid) as implicit GEMM via WMMA bf16.
//   C[pix=2048, cout] = sum_K  A[pix,K] * B[K,cout],  K = cin*9 + ky*3 + kx
//   one wave per 16x16 output tile, K staged 32-wide through LDS.
// ---------------------------------------------------------------------------
__global__ void conv_bn_wmma_kernel(const float* __restrict__ x,   // [32,256,8,8]
                                    const float* __restrict__ w,   // [Cout,256,3,3]
                                    const float* __restrict__ bn_g,
                                    const float* __restrict__ bn_b,
                                    const float* __restrict__ bn_m,
                                    const float* __restrict__ bn_v,
                                    float* __restrict__ y,         // [32,Cout,8,8]
                                    int Cout, int applySigmoid) {
  __shared__ unsigned short aT[16][32];   // [pixel-row][k]
  __shared__ unsigned short bT[16][32];   // [cout-col][k]  (k-contiguous)
  const int lane     = threadIdx.x;       // wave32
  const int pixBase  = blockIdx.x * 16;
  const int coutBase = blockIdx.y * 16;
  const int KTOT     = 256 * 9;

  v8f acc = {};
  for (int kb = 0; kb < KTOT; kb += 32) {
    // prefetch next weight block into cache (global_prefetch_b8)
    if (kb + 32 < KTOT)
      __builtin_prefetch(&w[(size_t)coutBase * KTOT + kb + 32], 0, 1);
    // stage A (im2col gather, zero pad)
    for (int idx = lane; idx < 512; idx += 32) {
      int row = idx >> 5, kk = idx & 31;
      int K = kb + kk;
      int cin = K / 9, kp = K % 9;
      int ky = kp / 3, kx = kp % 3;
      int pix = pixBase + row;
      int b = pix >> 6, yx = pix & 63;
      int oy = yx >> 3, ox = yx & 7;
      int iy = oy + ky - 1, ix = ox + kx - 1;
      float val = 0.0f;
      if (iy >= 0 && iy < 8 && ix >= 0 && ix < 8)
        val = x[((b * 256 + cin) * 8 + iy) * 8 + ix];
      aT[row][kk] = f32_to_bf16_bits(val);
    }
    // stage B (weights, k-major per output channel)
    for (int idx = lane; idx < 512; idx += 32) {
      int n = idx >> 5, kk = idx & 31;
      bT[n][kk] = f32_to_bf16_bits(w[(size_t)(coutBase + n) * KTOT + kb + kk]);
    }
    __syncthreads();

    // A fragment: lane g=hi half, m=row.  e<8 -> K=g*8+e ; e>=8 -> K=16+g*8+(e-8)
    const int g = lane >> 4, m = lane & 15;
    v16us au, bu;
#pragma unroll
    for (int e = 0; e < 16; ++e) {
      int Ka = ((e & 8) << 1) + g * 8 + (e & 7);
      au[e] = aT[m][Ka];
      bu[e] = bT[m][Ka];                  // B: same K as A, column n == m
    }
    v16bf af = __builtin_bit_cast(v16bf, au);
    v16bf bf = __builtin_bit_cast(v16bf, bu);
    acc = __builtin_amdgcn_wmma_f32_16x16x32_bf16(false, af, false, bf,
                                                  (short)0, acc, false, false);
    __syncthreads();
  }

  // epilogue: BN (+sigmoid).  C/D layout: N=lane&15, M=r + 8*(lane>>4)
  const int g = lane >> 4, n = lane & 15;
  const int cout = coutBase + n;
  float scale = bn_g[cout] * rsqrtf(bn_v[cout] + 1e-5f);
  float shift = bn_b[cout] - bn_m[cout] * scale;
#pragma unroll
  for (int rr = 0; rr < 8; ++rr) {
    int pix = pixBase + rr + 8 * g;
    int b = pix >> 6, yx = pix & 63;
    float val = acc[rr] * scale + shift;
    if (applySigmoid) val = 1.0f / (1.0f + expf(-val));
    y[((size_t)(b * Cout + cout) * 64) + yx] = val;
  }
}

// ---------------------------------------------------------------------------
// Kernel 2: votes  v[s,r,o,i,j] = sum_h pc[s,r,i,h] * W[r,o,h,j]
// WMMA f32 16x16x4:  A[(o,j),h] = W[r],  B[h,(s',i)] = pose fragments of
// 4 sites -> 16 columns.  One wave per (site-group, r).
// ---------------------------------------------------------------------------
#if __has_builtin(__builtin_amdgcn_wmma_f32_16x16x4_f32)
#define HAVE_WMMA_F32K4 1
#endif

__global__ void votes_wmma_kernel(const float* __restrict__ pose,   // [32,256,8,8]
                                  const float* __restrict__ W,      // [R,B,4,4]
                                  unsigned short* __restrict__ votes, // bf16 [S,R,B,16]
                                  int R, int Bcaps, int L, int OW,
                                  int ksize, int pad) {
  const int lane = threadIdx.x;
  const int sg   = blockIdx.x;       // group of 4 sites
  const int r    = blockIdx.y;       // vote row (kk_idx*16 + a_cap)
  const int g    = lane >> 4, nn = lane & 15;

  const int kk_idx = r >> 4;         // A == 16 input capsules
  const int a_cap  = r & 15;
  const int ky = kk_idx / ksize, kx = kk_idx % ksize;

  __builtin_prefetch(&W[((size_t)r * Bcaps) * 16], 0, 1);

  // B fragment: column (s', i), rows h = 2g, 2g+1
  const int sp = nn >> 2, i = nn & 3;
  const int site = sg * 4 + sp;
  const int b = site / L, pos = site % L;
  const int oy = pos / OW, ox = pos % OW;
  const int iy = oy + ky - pad, ix = ox + kx - pad;
  float b0 = 0.f, b1 = 0.f;
  if (iy >= 0 && iy < 8 && ix >= 0 && ix < 8) {
    const int chBase = a_cap * 16 + i * 4;
    const float* p = pose + ((size_t)(b * 256 + chBase) * 64) + iy * 8 + ix;
    b0 = p[(size_t)(2 * g) * 64];
    b1 = p[(size_t)(2 * g + 1) * 64];
  }
  v2f bfrag = {b0, b1};

  const int rowsTot = Bcaps * 4;                 // (o,j) rows
  const int nTiles  = (rowsTot + 15) >> 4;
  for (int t = 0; t < nTiles; ++t) {
    int mrow = t * 16 + nn;
    float a0 = 0.f, a1 = 0.f;
    if (mrow < rowsTot) {
      int o = mrow >> 2, j = mrow & 3;
      const float* wp = W + (((size_t)r * Bcaps + o) * 4) * 4 + j;
      a0 = wp[(2 * g) * 4];
      a1 = wp[(2 * g + 1) * 4];
    }
    v2f afrag = {a0, a1};
    v8f d = {};
#if defined(HAVE_WMMA_F32K4)
    d = __builtin_amdgcn_wmma_f32_16x16x4_f32(false, afrag, false, bfrag,
                                              (short)0, d, false, false);
#else
    // shuffle-based fallback (bit-equivalent result layout)
#pragma unroll
    for (int rr = 0; rr < 8; ++rr) {
      float accv = 0.f;
#pragma unroll
      for (int h = 0; h < 4; ++h) {
        int srcA = (rr + 8 * g) + 16 * (h >> 1);
        float av = (h & 1) ? __shfl(afrag.y, srcA, 32) : __shfl(afrag.x, srcA, 32);
        int srcB = nn + 16 * (h >> 1);
        float bv = (h & 1) ? __shfl(bfrag.y, srcB, 32) : __shfl(bfrag.x, srcB, 32);
        accv += av * bv;
      }
      d[rr] = accv;
    }
#endif
    // D layout: column n = (s',i) (this lane's), row m = t*16 + rr + 8g
#pragma unroll
    for (int rr = 0; rr < 8; ++rr) {
      int mr = t * 16 + rr + 8 * g;
      if (mr < rowsTot) {
        int o = mr >> 2, j = mr & 3;
        size_t off = (((size_t)site * R + r) * Bcaps + o) * 16 + i * 4 + j;
        votes[off] = f32_to_bf16_bits(d[rr]);
      }
    }
  }
}

// ---------------------------------------------------------------------------
// Kernel 3: EM routing, one workgroup (256 threads) per output site.
// Votes staged into LDS (bf16) via async-to-LDS; 3 iterations LDS-resident.
// Optional fused BN on the pose output.
// ---------------------------------------------------------------------------
__global__ void em_routing_kernel(const unsigned short* __restrict__ votes, // bf16 [S,R,B,16]
                                  const float* __restrict__ a_prev,        // [32,16,8,8]
                                  const float* __restrict__ beta_u,
                                  const float* __restrict__ beta_a,
                                  const float* __restrict__ bn_g,
                                  const float* __restrict__ bn_b,
                                  const float* __restrict__ bn_m,
                                  const float* __restrict__ bn_v,
                                  float* __restrict__ a_out_g,   // [32,B,OH,OW]
                                  float* __restrict__ pose_out_g,// [32,B*16,OH,OW] or null
                                  int R, int Bcaps, int L, int OW,
                                  int ksize, int pad) {
  extern __shared__ char smem[];
  unsigned short* vlds = (unsigned short*)smem;                       // R*B*16 bf16
  size_t voff = ((size_t)R * Bcaps * 16 * 2 + 15) & ~(size_t)15;
  float* a_in = (float*)(smem + voff);                                // R
  float* rbuf = a_in + R;                                             // R*B
  float* mu   = rbuf + (size_t)R * Bcaps;                             // B*16
  float* sig  = mu + Bcaps * 16;                                      // B*16
  float* rsum = sig + Bcaps * 16;                                     // B
  float* aout = rsum + Bcaps;                                         // B

  const int tid = threadIdx.x;
  const int s = blockIdx.x;
  const int b = s / L, pos = s % L;
  const int oy = pos / OW, ox = pos % OW;
  const int OH = L / OW;
  const float EPSc = 1e-8f;

  // ---- bulk-stage this site's votes into LDS --------------------------------
  const int nbytes = R * Bcaps * 32;                  // bf16 bytes
#if HAVE_ASYNC_LDS
  {
    // async DMA path: no VGPR staging, tracked by ASYNCcnt
    glb_v4i_ptr gsrc =
        (glb_v4i_ptr)(const void*)((const char*)votes + (size_t)s * nbytes);
    lds_v4i_ptr ldst = (lds_v4i_ptr)(void*)smem;
    const int nchunks = nbytes >> 4;                  // 16B per async op
    for (int c = tid; c < nchunks; c += blockDim.x)
      __builtin_amdgcn_global_load_async_to_lds_b128(gsrc + c, ldst + c, 0, 0);
  }
#else
  {
    const uint4* src = (const uint4*)((const char*)votes + (size_t)s * nbytes);
    uint4* dst = (uint4*)vlds;
    const int n16 = nbytes >> 4;
    for (int idx = tid; idx < n16; idx += blockDim.x) dst[idx] = src[idx];
  }
#endif
  // unfold activations (zero-padded) -- overlaps with the async copy
  for (int rr = tid; rr < R; rr += blockDim.x) {
    int kk_idx = rr >> 4, a_cap = rr & 15;
    int ky = kk_idx / ksize, kx = kk_idx % ksize;
    int iy = oy + ky - pad, ix = ox + kx - pad;
    float v = 0.f;
    if (iy >= 0 && iy < 8 && ix >= 0 && ix < 8)
      v = a_prev[((b * 16 + a_cap) * 8 + iy) * 8 + ix];
    a_in[rr] = v;
  }
  const float rinit = 1.0f / (float)Bcaps;
  for (int idx = tid; idx < R * Bcaps; idx += blockDim.x) rbuf[idx] = rinit;
#if HAVE_ASYNC_LDS
  wait_asynccnt_zero();
#endif
  __syncthreads();

  for (int it = 0; it < 3; ++it) {
    float lam = 1e-2f * (1.0f - powf(0.95f, (float)(it + 1)));
    // ---- M step: rw = normalize_o(r * a_in)
    if (tid < R) {
      float ai = a_in[tid];
      float tmp[16];
      float ssum = 0.f;
      for (int o = 0; o < Bcaps; ++o) {
        float wv = rbuf[tid * Bcaps + o] * ai;
        tmp[o] = wv; ssum += wv;
      }
      float inv = 1.0f / (ssum + EPSc);
      for (int o = 0; o < Bcaps; ++o) rbuf[tid * Bcaps + o] = tmp[o] * inv;
    }
    __syncthreads();
    if (tid < Bcaps) {
      float ssum = 0.f;
      for (int rr = 0; rr < R; ++rr) ssum += rbuf[rr * Bcaps + tid];
      rsum[tid] = ssum;
    }
    __syncthreads();
    {   // mu[o,p] = (sum_r rw*v) / (rsum+eps)
      int o = tid >> 4, p = tid & 15;
      if (o < Bcaps) {
        float num = 0.f;
        for (int rr = 0; rr < R; ++rr)
          num += rbuf[rr * Bcaps + o] *
                 bf16_bits_to_f32(vlds[((size_t)rr * Bcaps + o) * 16 + p]);
        mu[o * 16 + p] = num / (rsum[o] + EPSc);
      }
    }
    __syncthreads();
    {   // sig[o,p] = (sum_r rw*(v-mu)^2)/(rsum+eps) + eps
      int o = tid >> 4, p = tid & 15;
      if (o < Bcaps) {
        float m = mu[o * 16 + p];
        float num = 0.f;
        for (int rr = 0; rr < R; ++rr) {
          float d = bf16_bits_to_f32(vlds[((size_t)rr * Bcaps + o) * 16 + p]) - m;
          num += rbuf[rr * Bcaps + o] * d * d;
        }
        sig[o * 16 + p] = num / (rsum[o] + EPSc) + EPSc;
      }
    }
    __syncthreads();
    if (tid < Bcaps) {   // cost + activation
      float ssum = 0.f;
      for (int p = 0; p < 16; ++p)
        ssum += beta_u[tid] + 0.5f * logf(sig[tid * 16 + p]);
      float cost = ssum * rsum[tid];
      aout[tid] = 1.0f / (1.0f + expf(-lam * (beta_a[tid] - cost)));
    }
    __syncthreads();
    if (it < 2) {        // ---- E step
      for (int idx = tid; idx < R * Bcaps; idx += blockDim.x) {
        int rr = idx / Bcaps, o = idx % Bcaps;
        float ssum = 0.f;
        for (int p = 0; p < 16; ++p) {
          float sg_ = sig[o * 16 + p];
          float d = bf16_bits_to_f32(vlds[((size_t)rr * Bcaps + o) * 16 + p]) -
                    mu[o * 16 + p];
          ssum += -0.5f * d * d / sg_ - 0.5f * logf(sg_);
        }
        ssum -= 8.0f * 1.8378770664093454f;     // 16 * 0.5 * ln(2*pi)
        rbuf[idx] = ssum + logf(aout[o]);
      }
      __syncthreads();
      if (tid < R) {     // softmax over o
        float mx = -3.0e38f;
        for (int o = 0; o < Bcaps; ++o) mx = fmaxf(mx, rbuf[tid * Bcaps + o]);
        float se = 0.f;
        for (int o = 0; o < Bcaps; ++o) {
          float e = expf(rbuf[tid * Bcaps + o] - mx);
          rbuf[tid * Bcaps + o] = e; se += e;
        }
        float inv = 1.0f / se;
        for (int o = 0; o < Bcaps; ++o) rbuf[tid * Bcaps + o] *= inv;
      }
      __syncthreads();
    }
  }

  // outputs (optionally BN-fused pose)
  if (tid < Bcaps)
    a_out_g[(((size_t)b * Bcaps + tid) * OH + oy) * OW + ox] = aout[tid];
  if (pose_out_g != nullptr && tid < Bcaps * 16) {
    float val = mu[tid];
    if (bn_g != nullptr) {
      float scale = bn_g[tid] * rsqrtf(bn_v[tid] + 1e-5f);
      val = (val - bn_m[tid]) * scale + bn_b[tid];
    }
    pose_out_g[(((size_t)b * (Bcaps * 16) + tid) * OH + oy) * OW + ox] = val;
  }
}

// ---------------------------------------------------------------------------
// Kernel 4: spatial mean of fc activations -> [32,10]
// ---------------------------------------------------------------------------
__global__ void mean_kernel(const float* __restrict__ a_fc, float* __restrict__ out) {
  int idx = blockIdx.x * blockDim.x + threadIdx.x;
  if (idx < 320) {
    int b = idx / 10, o = idx % 10;
    float s = 0.f;
    for (int p = 0; p < 25; ++p) s += a_fc[((size_t)b * 10 + o) * 25 + p];
    out[idx] = s * (1.0f / 25.0f);
  }
}

// ---------------------------------------------------------------------------
static size_t em_shmem_bytes(int R, int B) {
  size_t s = ((size_t)R * B * 16 * 2 + 15) & ~(size_t)15; // votes bf16
  s += (size_t)R * 4;            // a_in
  s += (size_t)R * B * 4;        // rbuf
  s += (size_t)B * 16 * 4;       // mu
  s += (size_t)B * 16 * 4;       // sig
  s += (size_t)B * 4;            // rsum
  s += (size_t)B * 4;            // aout
  return s;
}

extern "C" void kernel_launch(void* const* d_in, const int* in_sizes, int n_in,
                              void* d_out, int out_size, void* d_ws, size_t ws_size,
                              hipStream_t stream) {
  (void)in_sizes; (void)n_in; (void)out_size; (void)ws_size;
  const float* x           = (const float*)d_in[0];
  const float* conv_a_w    = (const float*)d_in[1];
  const float* conv_pose_w = (const float*)d_in[2];
  const float* bn_a_g  = (const float*)d_in[3];
  const float* bn_a_b  = (const float*)d_in[4];
  const float* bn_a_m  = (const float*)d_in[5];
  const float* bn_a_v  = (const float*)d_in[6];
  const float* bn_p_g  = (const float*)d_in[7];
  const float* bn_p_b  = (const float*)d_in[8];
  const float* bn_p_m  = (const float*)d_in[9];
  const float* bn_p_v  = (const float*)d_in[10];
  const float* em0_W   = (const float*)d_in[11];
  const float* em0_bu  = (const float*)d_in[12];
  const float* em0_ba  = (const float*)d_in[13];
  const float* bn0_g   = (const float*)d_in[14];
  const float* bn0_b   = (const float*)d_in[15];
  const float* bn0_m   = (const float*)d_in[16];
  const float* bn0_v   = (const float*)d_in[17];
  const float* em1_W   = (const float*)d_in[18];
  const float* em1_bu  = (const float*)d_in[19];
  const float* em1_ba  = (const float*)d_in[20];
  const float* bn1_g   = (const float*)d_in[21];
  const float* bn1_b   = (const float*)d_in[22];
  const float* bn1_m   = (const float*)d_in[23];
  const float* bn1_v   = (const float*)d_in[24];
  const float* fc_W    = (const float*)d_in[25];
  const float* fc_bu   = (const float*)d_in[26];
  const float* fc_ba   = (const float*)d_in[27];

  // workspace carve-up
  char* ws = (char*)d_ws;
  size_t off = 0;
  auto carve = [&](size_t bytes) -> void* {
    void* p = ws + off;
    off = (off + bytes + 255) & ~(size_t)255;
    return p;
  };
  float* aA    = (float*)carve((size_t)32 * 16 * 64 * 4);          // 128 KB
  float* aB    = (float*)carve((size_t)32 * 16 * 64 * 4);
  float* poseA = (float*)carve((size_t)32 * 256 * 64 * 4);         // 2 MB
  float* poseB = (float*)carve((size_t)32 * 256 * 64 * 4);
  float* aFC   = (float*)carve((size_t)32 * 10 * 25 * 4);
  unsigned short* votes = (unsigned short*)carve((size_t)2048 * 144 * 256 * 2); // 151 MB bf16

  // allow >64KB dynamic LDS for the EM kernel (WGP has 320KB)
  (void)hipFuncSetAttribute((const void*)em_routing_kernel,
                            hipFuncAttributeMaxDynamicSharedMemorySize,
                            160 * 1024);

  const size_t shEM = em_shmem_bytes(144, 16);
  const size_t shFC = em_shmem_bytes(256, 10);

  // conv_a: sigmoid(BN(conv))      conv_pose: BN(conv)
  conv_bn_wmma_kernel<<<dim3(128, 1), 32, 0, stream>>>(
      x, conv_a_w, bn_a_g, bn_a_b, bn_a_m, bn_a_v, aA, 16, 1);
  conv_bn_wmma_kernel<<<dim3(128, 16), 32, 0, stream>>>(
      x, conv_pose_w, bn_p_g, bn_p_b, bn_p_m, bn_p_v, poseA, 256, 0);

  // EM layer 0 (k=3, pad=1, l=64, R=144, B=16) ; BN0 fused into pose output
  votes_wmma_kernel<<<dim3(512, 144), 32, 0, stream>>>(
      poseA, em0_W, votes, 144, 16, 64, 8, 3, 1);
  em_routing_kernel<<<2048, 256, shEM, stream>>>(
      votes, aA, em0_bu, em0_ba, bn0_g, bn0_b, bn0_m, bn0_v,
      aB, poseB, 144, 16, 64, 8, 3, 1);

  // EM layer 1 ; BN1 fused
  votes_wmma_kernel<<<dim3(512, 144), 32, 0, stream>>>(
      poseB, em1_W, votes, 144, 16, 64, 8, 3, 1);
  em_routing_kernel<<<2048, 256, shEM, stream>>>(
      votes, aB, em1_bu, em1_ba, bn1_g, bn1_b, bn1_m, bn1_v,
      aA, poseA, 144, 16, 64, 8, 3, 1);

  // fc "class capsules" layer (k=4, pad=0, l=25, R=256, B=10)
  votes_wmma_kernel<<<dim3(200, 256), 32, 0, stream>>>(
      poseA, fc_W, votes, 256, 10, 25, 5, 4, 0);
  em_routing_kernel<<<800, 256, shFC, stream>>>(
      votes, aA, fc_bu, fc_ba, nullptr, nullptr, nullptr, nullptr,
      aFC, nullptr, 256, 10, 25, 5, 4, 0);

  mean_kernel<<<1, 320, 0, stream>>>(aFC, (float*)d_out);
}
